// CausalSelfAttention_7559142441541
// MI455X (gfx1250) — matrix operand order repaired and checked by reading, not compiled, not run on hardware
//
#include <hip/hip_runtime.h>
#include <hip/hip_bf16.h>
#include <math.h>
#include <stdint.h>

#define TSEQ 1024
#define CDIM 1024
#define NB 4
#define NH 16
#define HD 64
#define MEMW 64

typedef __attribute__((ext_vector_type(16))) __bf16 v16bf;
typedef __attribute__((ext_vector_type(8)))  float  v8f;

#define NEG_INF (-__builtin_inff())

static __device__ __forceinline__ v8f wmma_bf16(v16bf a, v16bf b, v8f c) {
  return __builtin_amdgcn_wmma_f32_16x16x32_bf16(false, a, false, b, (short)0, c, false, false);
}

// ---- WMMA fragment loaders (wave32; contiguous k = 16*(lane/16) + element, same in A and B) ----

// A fragment 16x32 bf16 from row-major [m][k] bf16, row stride ld (elements).
static __device__ __forceinline__ v16bf load_A_bf16(const __bf16* p, int ld, int lane) {
  const int m = lane & 15, hi = lane >> 4;
  union { v16bf v; unsigned int u[8]; } a;
#pragma unroll
  for (int r = 0; r < 8; ++r) {
    const int kk = 2 * r + (hi << 4);
    a.u[r] = *(const unsigned int*)(p + (size_t)m * ld + kk);
  }
  return a.v;
}

// B fragment 32x16 where memory holds B transposed: row-major [n][k] bf16
// (K matrix stored [key][hd], or pre-transposed weights [N][K]; dword loads).
static __device__ __forceinline__ v16bf load_BT_bf16(const __bf16* p, int ld, int lane) {
  const int n = lane & 15, hi = lane >> 4;
  union { v16bf v; unsigned int u[8]; } b;
#pragma unroll
  for (int r = 0; r < 8; ++r) {
    const int kk = 2 * r + (hi << 4);
    b.u[r] = *(const unsigned int*)(p + (size_t)n * ld + kk);
  }
  return b.v;
}

// B fragment 32x16 of V rows [key][hd] with clamped (possibly OOB-padded) keys.
static __device__ __forceinline__ v16bf load_vB(const __bf16* vb, int key0, int d0, int lane) {
  const int n = lane & 15, hi = lane >> 4;
  union { v16bf v; __bf16 h[16]; } b;
#pragma unroll
  for (int r = 0; r < 8; ++r) {
    int k0 = key0 + 2 * r + (hi << 4);
    int k1 = k0 + 1;
    int k0c = k0 < 0 ? 0 : (k0 > TSEQ - 1 ? TSEQ - 1 : k0);
    int k1c = k1 < 0 ? 0 : (k1 > TSEQ - 1 ? TSEQ - 1 : k1);
    b.h[2 * r]     = vb[(size_t)k0c * HD + d0 + n];
    b.h[2 * r + 1] = vb[(size_t)k1c * HD + d0 + n];
  }
  return b.v;
}

// ---- 16-lane cross-lane reductions (rows of the C/D layout live in 16-lane halves) ----
static __device__ __forceinline__ float redmax16(float v) {
#pragma unroll
  for (int off = 1; off < 16; off <<= 1) v = fmaxf(v, __shfl_xor(v, off, 16));
  return v;
}
static __device__ __forceinline__ float redsum16(float v) {
#pragma unroll
  for (int off = 1; off < 16; off <<= 1) v += __shfl_xor(v, off, 16);
  return v;
}

// ================= Prologue A: x fp32 -> bf16, row-major copy (vectorized) ===============
__global__ __launch_bounds__(256) void cvt_x_kernel(const float* __restrict__ src,
                                                    __bf16* __restrict__ dst) {
  const size_t i = ((size_t)blockIdx.x * blockDim.x + threadIdx.x) * 8;
  float4 f0 = *(const float4*)(src + i);
  float4 f1 = *(const float4*)(src + i + 4);
  union { __bf16 h[8]; uint4 u; } o;
  o.h[0] = (__bf16)f0.x; o.h[1] = (__bf16)f0.y;
  o.h[2] = (__bf16)f0.z; o.h[3] = (__bf16)f0.w;
  o.h[4] = (__bf16)f1.x; o.h[5] = (__bf16)f1.y;
  o.h[6] = (__bf16)f1.z; o.h[7] = (__bf16)f1.w;
  *(uint4*)(dst + i) = o.u;
}

// ====== Prologue B: weight fp32 [rows][cols] -> bf16 transposed [cols][rows] =============
// Uses the gfx1250 async global->LDS DMA path (ASYNCcnt) to stage the fp32 tile:
// global_load_async_to_lds_b128 moves 16B/lane with no VGPR round-trip.
__global__ __launch_bounds__(256) void transpose_cvt_kernel(const float* __restrict__ src,
                                                            __bf16* __restrict__ dst,
                                                            int rows, int cols) {
  __shared__ float  ftile[32][32];
  __shared__ __bf16 tile[32][33];
  const int c0 = blockIdx.x << 5, r0 = blockIdx.y << 5;
  const int tx = threadIdx.x & 31, ty = threadIdx.x >> 5;   // 32 x 8

  {
    const int row  = threadIdx.x >> 3;          // 0..31
    const int col4 = (threadIdx.x & 7) << 2;    // 0,4,...,28 (16B aligned)
    const float* g = src + (size_t)(r0 + row) * cols + c0 + col4;
    unsigned lds = (unsigned)(uintptr_t)&ftile[row][col4];  // LDS byte offset (addr[31:0])
    asm volatile("global_load_async_to_lds_b128 %0, %1, off"
                 :: "v"(lds), "v"(g) : "memory");
    asm volatile("s_wait_asynccnt 0x0" ::: "memory");
  }
  __syncthreads();
#pragma unroll
  for (int i = 0; i < 32; i += 8)
    tile[ty + i][tx] = (__bf16)ftile[ty + i][tx];
  __syncthreads();
#pragma unroll
  for (int i = 0; i < 32; i += 8)
    dst[(size_t)(c0 + ty + i) * rows + r0 + tx] = tile[tx][ty + i];
}

// ================= Kernel 1: qkv = x @ W_attn, scatter to q/k/v [B,H,T,hd] bf16 ==========
// xb: bf16 [B*T][C]; WaT: bf16 [3C][C] (transposed weight, fragment-native).
__global__ __launch_bounds__(256) void qkv_gemm_kernel(const __bf16* __restrict__ xb,
                                                       const __bf16* __restrict__ WaT,
                                                       __bf16* __restrict__ q,
                                                       __bf16* __restrict__ k,
                                                       __bf16* __restrict__ v) {
  const int lane = threadIdx.x & 31;
  const int wave = blockIdx.x * (blockDim.x >> 5) + (threadIdx.x >> 5);
  const int NT = (3 * CDIM) >> 4;                 // 192 N-tiles
  const int mt = wave / NT;
  const int nt = wave - mt * NT;
  if (mt >= (NB * TSEQ) >> 4) return;

  v8f acc = {};
  const __bf16* ap = xb + (size_t)(mt << 4) * CDIM;
  const __bf16* bp = WaT + (size_t)(nt << 4) * CDIM;
  for (int kk = 0; kk < CDIM; kk += 32) {
    if (kk + 32 < CDIM) {
      __builtin_prefetch(ap + kk + 32, 0, 1);
      __builtin_prefetch(bp + kk + 32, 0, 1);
    }
    v16bf a = load_A_bf16(ap + kk, CDIM, lane);
    v16bf b = load_BT_bf16(bp + kk, CDIM, lane);
    acc = wmma_bf16(a, b, acc);
  }

  const int hi = lane >> 4, nn = lane & 15;
  const int col = (nt << 4) + nn;
  const int which = col >> 10;                    // 0:q 1:k 2:v
  const int cc = col & 1023;
  const int h = cc >> 6, d = cc & 63;
  __bf16* dst = which == 0 ? q : (which == 1 ? k : v);
#pragma unroll
  for (int r = 0; r < 8; ++r) {
    const int row = (mt << 4) + r + (hi << 3);    // b*T + t
    const int bb = row >> 10, t = row & 1023;
    dst[(((size_t)(bb * NH + h)) * TSEQ + t) * HD + d] = (__bf16)acc[r];
  }
}

// ================= Kernel 2: attention (full-causal att + windowed y-heads) ==============
__global__ __launch_bounds__(32) void attn_kernel(const __bf16* __restrict__ qg,
                                                  const __bf16* __restrict__ kg,
                                                  const __bf16* __restrict__ vg,
                                                  float* __restrict__ att,
                                                  __bf16* __restrict__ yh) {
  __shared__ float  sw[5][16][16];    // windowed raw scores
  __shared__ __bf16 pa[16][32];       // windowed probs, A-fragment staging

  const int lane = threadIdx.x & 31;
  const int hi = lane >> 4, nn = lane & 15;
  const int bh = blockIdx.x >> 6;     // b*NH + h
  const int it = blockIdx.x & 63;     // 16-row query tile
  const int b = bh >> 4, h = bh & 15;
  const int i0 = it << 4;

  const __bf16* qb = qg + ((size_t)bh * TSEQ + i0) * HD;
  const __bf16* kb = kg + (size_t)bh * TSEQ * HD;
  const __bf16* vb = vg + (size_t)bh * TSEQ * HD;

  const v16bf qa0 = load_A_bf16(qb, HD, lane);        // K = 0..31 of hd
  const v16bf qa1 = load_A_bf16(qb + 32, HD, lane);   // K = 32..63

  float rm[8], rs[8];
#pragma unroll
  for (int r = 0; r < 8; ++r) { rm[r] = NEG_INF; rs[r] = 0.f; }

  float* attrow = att + ((size_t)bh * TSEQ + i0) * TSEQ;

  // ---- Pass 1: causal scores -> att (raw), online (max,sum) per row ----
  for (int jt = 0; jt <= it; ++jt) {
    v16bf b0 = load_BT_bf16(kb + (size_t)(jt << 4) * HD, HD, lane);
    v16bf b1 = load_BT_bf16(kb + (size_t)(jt << 4) * HD + 32, HD, lane);
    v8f c = {};
    c = wmma_bf16(qa0, b0, c);
    c = wmma_bf16(qa1, b1, c);
    const int jcol = (jt << 4) + nn;
#pragma unroll
    for (int r = 0; r < 8; ++r) {
      const int irow = i0 + r + (hi << 3);
      float s = c[r] * 0.125f;                  // 1/sqrt(64)
      if (jcol > irow) s = NEG_INF;             // causal mask (diag block only)
      attrow[(size_t)(r + (hi << 3)) * TSEQ + jcol] = s;
      const float bm = redmax16(s);
      const float mn = fmaxf(rm[r], bm);
      const float es = redsum16(__expf(s - mn));
      rs[r] = rs[r] * __expf(rm[r] - mn) + es;
      rm[r] = mn;
    }
  }

  float rinv[8];
#pragma unroll
  for (int r = 0; r < 8; ++r) rinv[r] = 1.f / rs[r];

  // ---- Pass 2: normalize att (L2-hot re-read), zero the upper triangle ----
  for (int jt = 0; jt < 64; ++jt) {
    const int jcol = (jt << 4) + nn;
#pragma unroll
    for (int r = 0; r < 8; ++r) {
      const size_t idx = (size_t)(r + (hi << 3)) * TSEQ + jcol;
      float p = 0.f;
      if (jt <= it) p = __expf(attrow[idx] - rm[r]) * rinv[r];
      attrow[idx] = p;
    }
  }

  // ---- Window phase: recompute <=5 diagonal key tiles, band mask, softmax ----
  float wm[8], ws[8];
#pragma unroll
  for (int r = 0; r < 8; ++r) { wm[r] = NEG_INF; ws[r] = 0.f; }

  for (int jj = 0; jj < 5; ++jj) {
    const int jt = it - 4 + jj;
    if (jt < 0) {
#pragma unroll
      for (int r = 0; r < 8; ++r) sw[jj][r + (hi << 3)][nn] = NEG_INF;
      continue;
    }
    v16bf b0 = load_BT_bf16(kb + (size_t)(jt << 4) * HD, HD, lane);
    v16bf b1 = load_BT_bf16(kb + (size_t)(jt << 4) * HD + 32, HD, lane);
    v8f c = {};
    c = wmma_bf16(qa0, b0, c);
    c = wmma_bf16(qa1, b1, c);
    const int jcol = (jt << 4) + nn;
#pragma unroll
    for (int r = 0; r < 8; ++r) {
      const int irow = i0 + r + (hi << 3);
      float s = c[r] * 0.125f;
      const int dd = irow - jcol;
      if (dd < 0 || dd > MEMW) s = NEG_INF;     // band mask
      sw[jj][r + (hi << 3)][nn] = s;
      const float bm = redmax16(s);
      const float mn = fmaxf(wm[r], bm);
      const float es = redsum16(__expf(s - mn));
      ws[r] = ws[r] * __expf(wm[r] - mn) + es;
      wm[r] = mn;
    }
  }
  float wiv[8];
#pragma unroll
  for (int r = 0; r < 8; ++r) wiv[r] = 1.f / ws[r];

  // ---- y = P_window @ V : 3 pairs of key tiles (K=32 each), 4 hd subtiles ----
  v8f yacc[4] = {};
  for (int pair = 0; pair < 3; ++pair) {
    __syncthreads();
#pragma unroll
    for (int half = 0; half < 2; ++half) {
      const int jj = pair * 2 + half;
#pragma unroll
      for (int r = 0; r < 8; ++r) {
        const float s = (jj < 5) ? sw[jj][r + (hi << 3)][nn] : NEG_INF;
        const float p = __expf(s - wm[r]) * wiv[r];   // exp(-inf)=0 pads cleanly
        pa[r + (hi << 3)][(half << 4) + nn] = (__bf16)p;
      }
    }
    __syncthreads();
    const v16bf pf = load_A_bf16(&pa[0][0], 32, lane);
    const int key0 = (it - 4 + pair * 2) << 4;
#pragma unroll
    for (int s4 = 0; s4 < 4; ++s4) {
      v16bf vbf = load_vB(vb, key0, s4 << 4, lane);
      yacc[s4] = wmma_bf16(pf, vbf, yacc[s4]);
    }
  }

  // ---- store y-heads bf16 into [B,T,C] (undoes the head transpose) ----
#pragma unroll
  for (int s4 = 0; s4 < 4; ++s4) {
    const int col = (h << 6) + (s4 << 4) + nn;
#pragma unroll
    for (int r = 0; r < 8; ++r) {
      const int t = i0 + r + (hi << 3);
      yh[((size_t)(b * TSEQ + t)) * CDIM + col] = (__bf16)yacc[s4][r];
    }
  }
}

// ================= Kernel 3: att_std over batch (ddof=1) =================================
__global__ __launch_bounds__(256) void att_std_kernel(const float* __restrict__ att,
                                                      float* __restrict__ out) {
  const size_t e = (size_t)blockIdx.x * blockDim.x + threadIdx.x;
  const size_t stride = (size_t)NH * TSEQ * TSEQ;   // 16M
  const float a0 = att[e];
  const float a1 = att[e + stride];
  const float a2 = att[e + 2 * stride];
  const float a3 = att[e + 3 * stride];
  const float mu = 0.25f * (a0 + a1 + a2 + a3);
  const float d0 = a0 - mu, d1 = a1 - mu, d2 = a2 - mu, d3 = a3 - mu;
  out[e] = sqrtf((d0 * d0 + d1 * d1 + d2 * d2 + d3 * d3) * (1.0f / 3.0f));
}

// ================= Kernel 4: y = y_heads @ W_proj ========================================
// yh: bf16 [B*T][C]; WpT: bf16 [C][C] transposed weight.
__global__ __launch_bounds__(256) void proj_gemm_kernel(const __bf16* __restrict__ yh,
                                                        const __bf16* __restrict__ WpT,
                                                        float* __restrict__ y) {
  const int lane = threadIdx.x & 31;
  const int wave = blockIdx.x * (blockDim.x >> 5) + (threadIdx.x >> 5);
  const int NT = CDIM >> 4;                       // 64 N-tiles
  const int mt = wave / NT;
  const int nt = wave - mt * NT;
  if (mt >= (NB * TSEQ) >> 4) return;

  v8f acc = {};
  const __bf16* ap = yh + (size_t)(mt << 4) * CDIM;
  const __bf16* bp = WpT + (size_t)(nt << 4) * CDIM;
  for (int kk = 0; kk < CDIM; kk += 32) {
    if (kk + 32 < CDIM) {
      __builtin_prefetch(ap + kk + 32, 0, 1);
      __builtin_prefetch(bp + kk + 32, 0, 1);
    }
    v16bf a = load_A_bf16(ap + kk, CDIM, lane);
    v16bf b = load_BT_bf16(bp + kk, CDIM, lane);
    acc = wmma_bf16(a, b, acc);
  }
  const int hi = lane >> 4, nn = lane & 15;
#pragma unroll
  for (int r = 0; r < 8; ++r) {
    y[((size_t)((mt << 4) + r + (hi << 3))) * CDIM + (nt << 4) + nn] = acc[r];
  }
}

// ========================================================================================
extern "C" void kernel_launch(void* const* d_in, const int* in_sizes, int n_in,
                              void* d_out, int out_size, void* d_ws, size_t ws_size,
                              hipStream_t stream) {
  const float* x  = (const float*)d_in[0];   // [B,T,C]
  const float* Wa = (const float*)d_in[1];   // [C,3C]
  const float* Wp = (const float*)d_in[2];   // [C,C]

  float* y    = (float*)d_out;               // [B,T,C]        4,194,304
  float* att  = y + (size_t)NB * TSEQ * CDIM;              // [B,H,T,T] 67,108,864
  float* astd = att + (size_t)NB * NH * TSEQ * TSEQ;       // [H,T,T]   16,777,216

  const size_t qkv_elems = (size_t)NB * NH * TSEQ * HD;    // 4,194,304 each
  __bf16* q   = (__bf16*)d_ws;
  __bf16* k   = q + qkv_elems;
  __bf16* v   = k + qkv_elems;
  __bf16* yh  = v + qkv_elems;                             // [B,T,C] bf16
  __bf16* xb  = yh + (size_t)NB * TSEQ * CDIM;             // [B*T,C] bf16
  __bf16* WaT = xb + (size_t)NB * TSEQ * CDIM;             // [3C,C]  bf16
  __bf16* WpT = WaT + (size_t)3 * CDIM * CDIM;             // [C,C]   bf16

  // 0) one-time precision/layout conversion (removes all cvt + strided loads from GEMMs)
  cvt_x_kernel<<<2048, 256, 0, stream>>>(x, xb);                       // 4M elems / 8
  transpose_cvt_kernel<<<dim3(96, 32), 256, 0, stream>>>(Wa, WaT, CDIM, 3 * CDIM);
  transpose_cvt_kernel<<<dim3(32, 32), 256, 0, stream>>>(Wp, WpT, CDIM, CDIM);

  // 1) qkv GEMM: 256 M-tiles x 192 N-tiles = 49152 waves, 8 waves/block
  qkv_gemm_kernel<<<6144, 256, 0, stream>>>(xb, WaT, q, k, v);

  // 2) attention: one wave per (b,h, 16-row q-tile) = 4*16*64 = 4096 blocks
  attn_kernel<<<4096, 32, 0, stream>>>(q, k, v, att, yh);

  // 3) att_std: 16,777,216 outputs
  att_std_kernel<<<65536, 256, 0, stream>>>(att, astd);

  // 4) projection GEMM: 256 x 64 = 16384 waves, 8 waves/block
  proj_gemm_kernel<<<2048, 256, 0, stream>>>(yh, WpT, y);
}
